// LSTM_autoencoder_45114336477370
// MI455X (gfx1250) — matrix operand order repaired and checked by reading, not compiled, not run on hardware
//
#include <hip/hip_runtime.h>
#include <hip/hip_bf16.h>

// ---------------------------------------------------------------------------
// LSTM autoencoder for MI455X (gfx1250, wave32, WMMA 16x16x32 f16).
//   T=512, B=256, layer dims: 64 -> 128 -> 32(tanh) -> 128 -> 64
//
// Strategy:
//   1) Hoist input projection xg = x * Wih^T (biases added in the scan) out of the scan as a
//      fully parallel WMMA GEMM over T*B rows (saturates all WGPs), storing
//      xg transposed [T][4H][B] in f16 so the scan can load it directly into
//      the WMMA accumulator layout (8 consecutive batch rows per lane).
//   2) Batch-parallel recurrent scan: 16 workgroups x 16 batch rows, Whh
//      fragments resident in VGPRs, cell state in accumulator registers,
//      h exchanged through double-buffered LDS (ds_load_b128 / b16 stores).
//   3) If ws_size is too small for the xg buffer, fall back to the fused
//      variant that does the input GEMM inside the scan (decided once,
//      deterministically, from ws_size).
// ---------------------------------------------------------------------------

typedef __attribute__((ext_vector_type(16))) _Float16 v16h;
typedef __attribute__((ext_vector_type(8)))  _Float16 v8h;
typedef __attribute__((ext_vector_type(4)))  _Float16 v4h;
typedef __attribute__((ext_vector_type(8)))  float    v8f;
typedef __attribute__((ext_vector_type(4)))  float    v4f;

static constexpr int Tt = 512;   // sequence length
static constexpr int Bb = 256;   // batch

__device__ __forceinline__ float fsig(float x) {
    // sigmoid via v_exp_f32 + v_rcp_f32 (transcendental pipe, co-executes with WMMA)
    return __builtin_amdgcn_rcpf(1.0f + __expf(-x));
}
__device__ __forceinline__ float ftanh(float x) {
    return 2.0f * fsig(2.0f * x) - 1.0f;
}

// ---------------------------------------------------------------------------
// fp32 -> fp16 convert (vectorized, grid-stride)
// ---------------------------------------------------------------------------
__global__ void cvt_f32_to_f16(const float* __restrict__ in,
                               _Float16* __restrict__ out, int n4) {
    int i = blockIdx.x * blockDim.x + threadIdx.x;
    int stride = gridDim.x * blockDim.x;
    for (; i < n4; i += stride) {
        v4f v = *(const v4f*)(in + 4 * (size_t)i);
        v4h h;
        h[0] = (_Float16)v[0]; h[1] = (_Float16)v[1];
        h[2] = (_Float16)v[2]; h[3] = (_Float16)v[3];
        *(v4h*)(out + 4 * (size_t)i) = h;
    }
}

// ---------------------------------------------------------------------------
// Input projection GEMM:  xgT[t][col][b] = sum_k x[t][b][k] * w[col][k]
//   x   : [T*B, K]   f16
//   w16 : [N4H, K]   f16 (preconverted)
//   xgT : [T][N4H][B] f16 (transposed so the scan loads accumulator-layout)
// One 16(col) x 16(batch) WMMA tile per wave, 8 waves per block.
// ---------------------------------------------------------------------------
template <int K, int N4H>
__global__ __launch_bounds__(256) void xproj(
    const _Float16* __restrict__ xin,
    const _Float16* __restrict__ w16,
    _Float16* __restrict__ xgT)
{
    constexpr int NKT = K / 32;
    constexpr int CT  = N4H / 16;          // column tiles
    const int lane = threadIdx.x & 31;
    const int wv   = threadIdx.x >> 5;
    const int n    = lane & 15;
    const int hi   = lane >> 4;
    const int m    = n;

    const int tile = blockIdx.x * 8 + wv;  // (t * CT + ct) * 16 + bt
    const int bt   = tile & 15;
    const int rest = tile >> 4;
    const int ct   = rest % CT;
    const int t    = rest / CT;
    const int b0   = bt * 16;
    const int col0 = ct * 16;

    // A fragments: weight rows (col0+m), contiguous k = 16*hi + s (same as B)
    v16h af[NKT];
#pragma unroll
    for (int kt = 0; kt < NKT; kt++) {
        const _Float16* wr = w16 + (size_t)(col0 + m) * K + kt * 32 + hi * 8;
        af[kt] = *(const v16h*)(wr + hi * 8);
    }
    // B fragments: x rows; K is contiguous per lane -> single 32B load
    const _Float16* xr = xin + ((size_t)t * Bb + b0 + n) * K + hi * 16;
    v8f acc = {};
#pragma unroll
    for (int kt = 0; kt < NKT; kt++) {
        v16h bf = *(const v16h*)(xr + kt * 32);
        acc = __builtin_amdgcn_wmma_f32_16x16x32_f16(
                  false, af[kt], false, bf, (short)0, acc, false, false);
    }
    // store D: element r -> (col0 + r + 8*hi, b0 + n)
#pragma unroll
    for (int r = 0; r < 8; r++) {
        xgT[((size_t)t * N4H + col0 + r + 8 * hi) * Bb + b0 + n] = (_Float16)acc[r];
    }
}

// ---------------------------------------------------------------------------
// Recurrent scan with hoisted input projection.
//   H : hidden width; block has H/16 waves; wave w owns hidden cols [16w,16w+16)
//   of all four gates -> i/f/g/o combine is wave-local, cell state stays in a
//   v8f (WMMA C/D layout) for the entire scan.
// ---------------------------------------------------------------------------
template <int H, bool OUT_TANH, bool OUT_F32>
__global__ __launch_bounds__(32 * (H / 16)) void lstm_scan_h(
    const _Float16* __restrict__ xgT,   // [T][4H][B] f16 (pre-projected input)
    const float* __restrict__ w_hh,     // [4H, H]
    const float* __restrict__ b_ih,     // [4H]
    const float* __restrict__ b_hh,     // [4H]
    _Float16* __restrict__ hout_h,      // [T*B, H] f16   (used if !OUT_F32)
    float* __restrict__ hout_f)         // [T*B, H] f32   (used if  OUT_F32)
{
    constexpr int NKT_H = H / 32;
    const int lane = threadIdx.x & 31;
    const int wv   = threadIdx.x >> 5;
    const int b0   = blockIdx.x * 16;
    const int n    = lane & 15;
    const int hi   = lane >> 4;
    const int m    = n;

    __shared__ __align__(16) _Float16 hbuf[2][16][H];
    for (int idx = threadIdx.x; idx < 16 * H; idx += blockDim.x)
        hbuf[0][idx / H], hbuf[0][idx / H][idx & (H - 1)] = (_Float16)0.0f;

    // preload Whh fragments (resident in VGPRs for whole scan) + bias
    const int kbase = hi * 8;
    v16h Bhh[4][NKT_H];
    float bias[4];
#pragma unroll
    for (int q = 0; q < 4; q++) {
        const int col = q * H + wv * 16 + n;
        bias[q] = b_ih[col] + b_hh[col];
#pragma unroll
        for (int kt = 0; kt < NKT_H; kt++) {
            const float* src = w_hh + (size_t)col * H + kt * 32 + kbase;
            v16h tv;
#pragma unroll
            for (int e = 0; e < 16; e++) tv[e] = (_Float16)src[e < 8 ? e : e + 8];
            Bhh[q][kt] = tv;
        }
    }

    // per-gate xg pointers: fixed column (q*H + 16wv + n), 8 batch rows/lane
    const _Float16* xgp[4];
#pragma unroll
    for (int q = 0; q < 4; q++)
        xgp[q] = xgT + ((size_t)q * H + wv * 16 + n) * Bb + b0 + 8 * hi;
    constexpr size_t XG_STEP = (size_t)4 * H * Bb;   // halves per timestep

    v8f c = {};   // cell state, C/D layout: elem r -> (M = r + 8*hi, N = n)

    __syncthreads();  // h buffer 0 zeroed

    for (int t = 0; t < Tt; t++) {
        // gate init: bias + hoisted input projection (one b128 load per gate)
        v8f acc[4];
#pragma unroll
        for (int q = 0; q < 4; q++) {
            v8h xg = *(const v8h*)(xgp[q] + (size_t)t * XG_STEP);
            float bq = bias[q];
            v8f a;
#pragma unroll
            for (int r = 0; r < 8; r++) a[r] = bq + (float)xg[r];
            acc[q] = a;
        }
        if (t + 1 < Tt) {
            __builtin_prefetch(xgp[0] + (size_t)(t + 1) * XG_STEP, 0, 0);
            __builtin_prefetch(xgp[2] + (size_t)(t + 1) * XG_STEP, 0, 0);
        }

        // A fragments of h_{t-1} from LDS (ds_load_b128 pairs)
        const _Float16* hrow = &hbuf[t & 1][m][0];
        v16h ah[NKT_H];
#pragma unroll
        for (int kt = 0; kt < NKT_H; kt++) {
            const int off = kt * 32 + hi * 8;
            v8h lo = *(const v8h*)(hrow + off);
            v8h hb = *(const v8h*)(hrow + off + 16);
            ah[kt] = __builtin_shufflevector(lo, hb, 0, 1, 2, 3, 4, 5, 6, 7,
                                             8, 9, 10, 11, 12, 13, 14, 15);
        }

        // recurrent GEMM: 4 independent WMMA chains of length H/32
#pragma unroll
        for (int q = 0; q < 4; q++) {
#pragma unroll
            for (int kt = 0; kt < NKT_H; kt++)
                acc[q] = __builtin_amdgcn_wmma_f32_16x16x32_f16(
                             false, ah[kt], false, Bhh[q][kt], (short)0, acc[q],
                             false, false);
        }

        // elementwise cell update (gate order i,f,g,o)
        _Float16* hdst = &hbuf[(t + 1) & 1][0][0];
#pragma unroll
        for (int r = 0; r < 8; r++) {
            float ig = fsig(acc[0][r]);
            float fg = fsig(acc[1][r]);
            float gg = ftanh(acc[2][r]);
            float og = fsig(acc[3][r]);
            float cr = fg * c[r] + ig * gg;
            c[r] = cr;
            float hr = og * ftanh(cr);

            const int row = r + 8 * hi;
            const int col = wv * 16 + n;
            hdst[row * H + col] = (_Float16)hr;
            if (OUT_F32) {
                hout_f[((size_t)t * Bb + b0 + row) * H + col] = hr;
            } else {
                float ov = OUT_TANH ? ftanh(hr) : hr;
                hout_h[((size_t)t * Bb + b0 + row) * H + col] = (_Float16)ov;
            }
        }
        __syncthreads();
    }
}

// ---------------------------------------------------------------------------
// Fallback: fused scan (input GEMM inside the loop) — used if ws is small.
// ---------------------------------------------------------------------------
template <int KIN, int H, bool OUT_TANH, bool OUT_F32>
__global__ __launch_bounds__(32 * (H / 16)) void lstm_scan_fused(
    const _Float16* __restrict__ xin,
    const float* __restrict__ w_ih,
    const float* __restrict__ w_hh,
    const float* __restrict__ b_ih,
    const float* __restrict__ b_hh,
    _Float16* __restrict__ hout_h,
    float* __restrict__ hout_f)
{
    constexpr int NKT_IN = KIN / 32;
    constexpr int NKT_H  = H / 32;
    const int lane = threadIdx.x & 31;
    const int wv   = threadIdx.x >> 5;
    const int b0   = blockIdx.x * 16;
    const int n    = lane & 15;
    const int hi   = lane >> 4;
    const int m    = n;

    __shared__ __align__(16) _Float16 hbuf[2][16][H];
    for (int idx = threadIdx.x; idx < 16 * H; idx += blockDim.x)
        hbuf[0][idx / H][idx & (H - 1)] = (_Float16)0.0f;

    const int kbase = hi * 8;
    v16h Bih[4][NKT_IN];
    v16h Bhh[4][NKT_H];
    float bias[4];
#pragma unroll
    for (int q = 0; q < 4; q++) {
        const int col = q * H + wv * 16 + n;
        bias[q] = b_ih[col] + b_hh[col];
#pragma unroll
        for (int kt = 0; kt < NKT_IN; kt++) {
            const float* src = w_ih + (size_t)col * KIN + kt * 32 + kbase;
            v16h tv;
#pragma unroll
            for (int e = 0; e < 16; e++) tv[e] = (_Float16)src[e < 8 ? e : e + 8];
            Bih[q][kt] = tv;
        }
#pragma unroll
        for (int kt = 0; kt < NKT_H; kt++) {
            const float* src = w_hh + (size_t)col * H + kt * 32 + kbase;
            v16h tv;
#pragma unroll
            for (int e = 0; e < 16; e++) tv[e] = (_Float16)src[e < 8 ? e : e + 8];
            Bhh[q][kt] = tv;
        }
    }

    v8f c = {};
    __syncthreads();

    for (int t = 0; t < Tt; t++) {
        const _Float16* xrow = xin + ((size_t)t * Bb + b0 + m) * KIN;
        v16h ax[NKT_IN];
#pragma unroll
        for (int kt = 0; kt < NKT_IN; kt++) {
            const int off = kt * 32 + hi * 8;
            v8h lo = *(const v8h*)(xrow + off);
            v8h hb = *(const v8h*)(xrow + off + 16);
            ax[kt] = __builtin_shufflevector(lo, hb, 0, 1, 2, 3, 4, 5, 6, 7,
                                             8, 9, 10, 11, 12, 13, 14, 15);
        }
        if (t + 1 < Tt)
            __builtin_prefetch(xrow + (size_t)Bb * KIN, 0, 0);

        const _Float16* hrow = &hbuf[t & 1][m][0];
        v16h ah[NKT_H];
#pragma unroll
        for (int kt = 0; kt < NKT_H; kt++) {
            const int off = kt * 32 + hi * 8;
            v8h lo = *(const v8h*)(hrow + off);
            v8h hb = *(const v8h*)(hrow + off + 16);
            ah[kt] = __builtin_shufflevector(lo, hb, 0, 1, 2, 3, 4, 5, 6, 7,
                                             8, 9, 10, 11, 12, 13, 14, 15);
        }

        v8f acc[4];
#pragma unroll
        for (int q = 0; q < 4; q++) {
            float bq = bias[q];
            v8f a = {bq, bq, bq, bq, bq, bq, bq, bq};
#pragma unroll
            for (int kt = 0; kt < NKT_IN; kt++)
                a = __builtin_amdgcn_wmma_f32_16x16x32_f16(
                        false, ax[kt], false, Bih[q][kt], (short)0, a, false, false);
#pragma unroll
            for (int kt = 0; kt < NKT_H; kt++)
                a = __builtin_amdgcn_wmma_f32_16x16x32_f16(
                        false, ah[kt], false, Bhh[q][kt], (short)0, a, false, false);
            acc[q] = a;
        }

        _Float16* hdst = &hbuf[(t + 1) & 1][0][0];
#pragma unroll
        for (int r = 0; r < 8; r++) {
            float ig = fsig(acc[0][r]);
            float fg = fsig(acc[1][r]);
            float gg = ftanh(acc[2][r]);
            float og = fsig(acc[3][r]);
            float cr = fg * c[r] + ig * gg;
            c[r] = cr;
            float hr = og * ftanh(cr);

            const int row = r + 8 * hi;
            const int col = wv * 16 + n;
            hdst[row * H + col] = (_Float16)hr;
            if (OUT_F32) {
                hout_f[((size_t)t * Bb + b0 + row) * H + col] = hr;
            } else {
                float ov = OUT_TANH ? ftanh(hr) : hr;
                hout_h[((size_t)t * Bb + b0 + row) * H + col] = (_Float16)ov;
            }
        }
        __syncthreads();
    }
}

// ---------------------------------------------------------------------------
// host side
// ---------------------------------------------------------------------------
extern "C" void kernel_launch(void* const* d_in, const int* in_sizes, int n_in,
                              void* d_out, int out_size, void* d_ws, size_t ws_size,
                              hipStream_t stream) {
    (void)in_sizes; (void)n_in; (void)out_size;

    const float* X     = (const float*)d_in[0];
    const float* w1_ih = (const float*)d_in[1];
    const float* w1_hh = (const float*)d_in[2];
    const float* b1_ih = (const float*)d_in[3];
    const float* b1_hh = (const float*)d_in[4];
    const float* w2_ih = (const float*)d_in[5];
    const float* w2_hh = (const float*)d_in[6];
    const float* b2_ih = (const float*)d_in[7];
    const float* b2_hh = (const float*)d_in[8];
    const float* w3_ih = (const float*)d_in[9];
    const float* w3_hh = (const float*)d_in[10];
    const float* b3_ih = (const float*)d_in[11];
    const float* b3_hh = (const float*)d_in[12];
    const float* w4_ih = (const float*)d_in[13];
    const float* w4_hh = (const float*)d_in[14];
    const float* b4_ih = (const float*)d_in[15];
    const float* b4_hh = (const float*)d_in[16];

    const size_t TB = (size_t)Tt * Bb;
    float* out = (float*)d_out;
    const int NWG = Bb / 16;

    // workspace layout (halves)
    _Float16* x16 = (_Float16*)d_ws;          // [T*B, 64]
    _Float16* h1  = x16 + TB * 64;            // [T*B, 128]
    _Float16* h2  = h1  + TB * 128;           // [T*B, 32]
    _Float16* h3  = h2  + TB * 32;            // [T*B, 128]
    _Float16* w16a = h3 + TB * 128;           // 512*64
    _Float16* w16b = w16a + 512 * 64;         // 128*128
    _Float16* w16c = w16b + 128 * 128;        // 512*32
    _Float16* w16d = w16c + 512 * 32;         // 256*128
    _Float16* xg   = w16d + 256 * 128;        // [T][512][B] max
    const size_t need_hoist =
        ((size_t)(xg - x16) + TB * 512) * sizeof(_Float16);

    auto cvt = [&](const float* src, _Float16* dst, size_t nelem) {
        int n4 = (int)(nelem / 4);
        int blocks = (n4 + 255) / 256;
        if (blocks > 4096) blocks = 4096;
        cvt_f32_to_f16<<<blocks, 256, 0, stream>>>(src, dst, n4);
    };

    cvt(X, x16, TB * 64);

    if (ws_size >= need_hoist) {
        // ------ hoisted path: parallel xg GEMM + short-critical-path scans
        cvt(w1_ih, w16a, 512 * 64);
        cvt(w2_ih, w16b, 128 * 128);
        cvt(w3_ih, w16c, 512 * 32);
        cvt(w4_ih, w16d, 256 * 128);

        // en1: 64 -> 128
        xproj<64, 512><<<Tt * (512 / 16) * (Bb / 16) / 8, 256, 0, stream>>>(x16, w16a, xg);
        lstm_scan_h<128, false, false><<<NWG, 256, 0, stream>>>(
            xg, w1_hh, b1_ih, b1_hh, h1, nullptr);
        // en2: 128 -> 32 (+ bottleneck tanh fused into output store)
        xproj<128, 128><<<Tt * (128 / 16) * (Bb / 16) / 8, 256, 0, stream>>>(h1, w16b, xg);
        lstm_scan_h<32, true, false><<<NWG, 64, 0, stream>>>(
            xg, w2_hh, b2_ih, b2_hh, h2, nullptr);
        // de1: 32 -> 128
        xproj<32, 512><<<Tt * (512 / 16) * (Bb / 16) / 8, 256, 0, stream>>>(h2, w16c, xg);
        lstm_scan_h<128, false, false><<<NWG, 256, 0, stream>>>(
            xg, w3_hh, b3_ih, b3_hh, h3, nullptr);
        // de2: 128 -> 64, f32 straight to d_out
        xproj<128, 256><<<Tt * (256 / 16) * (Bb / 16) / 8, 256, 0, stream>>>(h3, w16d, xg);
        lstm_scan_h<64, false, true><<<NWG, 128, 0, stream>>>(
            xg, w4_hh, b4_ih, b4_hh, nullptr, out);
    } else {
        // ------ fused fallback (smaller workspace)
        lstm_scan_fused<64, 128, false, false><<<NWG, 256, 0, stream>>>(
            x16, w1_ih, w1_hh, b1_ih, b1_hh, h1, nullptr);
        lstm_scan_fused<128, 32, true, false><<<NWG, 64, 0, stream>>>(
            h1, w2_ih, w2_hh, b2_ih, b2_hh, h2, nullptr);
        lstm_scan_fused<32, 128, false, false><<<NWG, 256, 0, stream>>>(
            h2, w3_ih, w3_hh, b3_ih, b3_hh, h3, nullptr);
        lstm_scan_fused<128, 64, false, true><<<NWG, 128, 0, stream>>>(
            h3, w4_ih, w4_hh, b4_ih, b4_hh, nullptr, out);
    }
}
